// GRU_44770739093772
// MI455X (gfx1250) — matrix operand order repaired and checked by reading, not compiled, not run on hardware
//
#include <hip/hip_runtime.h>
#include <math.h>

#define B_ 64
#define S_ 512
#define I_ 512
#define H_ 1024

#define NTB 16              // N columns per scan block
#define NBLK (H_ / NTB)     // 64 persistent blocks
#define LDW 1032            // padded LDS row stride (bf16 elems): 516 dwords -> bank = 4*n

typedef __bf16 bf16_t;
typedef bf16_t v16bf __attribute__((ext_vector_type(16)));
typedef bf16_t v8bf  __attribute__((ext_vector_type(8)));
typedef float  v8f   __attribute__((ext_vector_type(8)));

// ---------- fragment loaders (wave32 CDNA5 WMMA layouts) ----------
// A 16x32 bf16: lane = khalf*16 + m ; elems 0..15 -> K = khalf*16..+15.
// p points at (row m, k0+khalf*16).
__device__ __forceinline__ v16bf load_frag_a(const bf16_t* p) {
    v8bf lo = *(const v8bf*)(p);
    v8bf hi = *(const v8bf*)(p + 8);
    return __builtin_shufflevector(lo, hi, 0,1,2,3,4,5,6,7,8,9,10,11,12,13,14,15);
}
// B 32x16 bf16 (B[k,n] = W[n,k]): lane = khalf*16 + n; 16 consecutive k at
// fixed n.  p points at (row n, k0+khalf*16); 16B-aligned halves.
__device__ __forceinline__ v16bf load_frag_b16(const bf16_t* p) {
    v8bf lo = *(const v8bf*)(p);
    v8bf hi = *(const v8bf*)(p + 8);
    return __builtin_shufflevector(lo, hi, 0,1,2,3,4,5,6,7,8,9,10,11,12,13,14,15);
}

__device__ __forceinline__ v8f wmma_bf16(v16bf a, v16bf b, v8f c) {
    return __builtin_amdgcn_wmma_f32_16x16x32_bf16(false, a, false, b,
                                                   (short)0, c, false, false);
}

__device__ __forceinline__ float sigmoidf_(float x) {
    return 1.0f / (1.0f + __expf(-x));
}

// ---------- prep: f32 -> bf16 cast ----------
__global__ void cvt_f32_to_bf16(const float* __restrict__ in,
                                bf16_t* __restrict__ out, int n) {
    int i = blockIdx.x * blockDim.x + threadIdx.x;
    if (i < n) out[i] = (bf16_t)in[i];
}

// ---------- prep: initialize h (f32 + bf16 copies) ----------
__global__ void init_h(const float* __restrict__ h0,
                       float* __restrict__ hf, bf16_t* __restrict__ hb, int n) {
    int i = blockIdx.x * blockDim.x + threadIdx.x;
    if (i < n) {
        float v = h0[i];
        hf[i] = v;
        hb[i] = (bf16_t)v;
    }
}

// ---------- prep: zero the inter-block barrier counters ----------
__global__ void zero_u32(unsigned int* __restrict__ p, int n) {
    int i = blockIdx.x * blockDim.x + threadIdx.x;
    if (i < n) p[i] = 0u;
}

// ---------- phase 1: X-gates = inputs @ W_i{r,z,n}^T + b_i{r,z,n} ----------
// M = S*B = 32768 (m = s*B + b), N = H, K = I.  Output bf16 [S*B, H] per gate.
// Block: 256 thr (8 waves), tile M=32 x N=64.  wave: mt = w&1, nt = w>>1.
__global__ void gate_gemm(const bf16_t* __restrict__ inb,   // [B,S,I] bf16
                          const bf16_t* __restrict__ wir,
                          const bf16_t* __restrict__ wiz,
                          const bf16_t* __restrict__ win,
                          const float* __restrict__ bir,
                          const float* __restrict__ biz,
                          const float* __restrict__ bin,
                          bf16_t* __restrict__ xr,
                          bf16_t* __restrict__ xz,
                          bf16_t* __restrict__ xn) {
    const int lane  = threadIdx.x & 31;
    const int wave  = threadIdx.x >> 5;
    const int mt    = wave & 1;
    const int nt    = wave >> 1;
    const int m0    = blockIdx.y * 32;            // global M base (rows m = s*B+b)
    const int s     = m0 >> 6;                    // 32-row tile stays in one s
    const int b0    = (m0 & 63) + mt * 16;
    const int n0    = blockIdx.x * 64 + nt * 16;
    const int khalf = lane >> 4;
    const int l16   = lane & 15;
    const int nrow  = n0 + l16;

    const bf16_t* arow = inb + ((size_t)(b0 + l16) * S_ + s) * I_ + khalf * 16;
    const bf16_t* pr   = wir + (size_t)nrow * I_ + khalf * 16;
    const bf16_t* pz   = wiz + (size_t)nrow * I_ + khalf * 16;
    const bf16_t* pn   = win + (size_t)nrow * I_ + khalf * 16;

    v8f accr = {}, accz = {}, accn = {};
    for (int k0 = 0; k0 < I_; k0 += 32) {
        v16bf a = load_frag_a(arow + k0);
        accr = wmma_bf16(a, load_frag_b16(pr + k0), accr);
        accz = wmma_bf16(a, load_frag_b16(pz + k0), accz);
        accn = wmma_bf16(a, load_frag_b16(pn + k0), accn);
    }

    const float vbr = bir[nrow], vbz = biz[nrow], vbn = bin[nrow];
    const int mbase = m0 + mt * 16 + 8 * khalf;
#pragma unroll
    for (int v = 0; v < 8; ++v) {
        size_t idx = (size_t)(mbase + v) * H_ + nrow;
        xr[idx] = (bf16_t)(accr[v] + vbr);
        xz[idx] = (bf16_t)(accz[v] + vbz);
        xn[idx] = (bf16_t)(accn[v] + vbn);
    }
}

// ---------- phase 2: persistent GRU scan ----------
// 64 blocks x 128 thr (4 waves).  Block owns N cols [n0, n0+16); wave w owns
// M tile [16w, 16w+16).  W_h slices (3 x 16 x 1024 bf16, padded ~96KB) staged
// into LDS once, reused for all 512 steps.  Per-step device-wide sync via
// atomic arrive counters (one per step) living in the workspace.
__global__ void __launch_bounds__(128)
gru_scan(const bf16_t* __restrict__ whr, const bf16_t* __restrict__ whz,
         const bf16_t* __restrict__ whn,
         const float* __restrict__ bhr, const float* __restrict__ bhz,
         const float* __restrict__ bhn,
         const bf16_t* __restrict__ xr, const bf16_t* __restrict__ xz,
         const bf16_t* __restrict__ xn,
         bf16_t* __restrict__ hbuf0, bf16_t* __restrict__ hbuf1,
         float* __restrict__ hf0, float* __restrict__ hf1,
         float* __restrict__ out, unsigned int* __restrict__ syncc) {
    __shared__ bf16_t wlds[3 * NTB * LDW];   // ~96.8 KB of the 320 KB WGP LDS

    const int lane  = threadIdx.x & 31;
    const int wave  = threadIdx.x >> 5;      // 4 waves = 4 M tiles
    const int khalf = lane >> 4;
    const int l16   = lane & 15;
    const int n0    = blockIdx.x * NTB;
    const int nrow  = n0 + l16;
    const int m0    = wave * 16;

    // ---- stage this block's recurrent-weight slices into LDS (once) ----
#pragma unroll
    for (int g = 0; g < 3; ++g) {
        const bf16_t* wsrc = (g == 0) ? whr : (g == 1) ? whz : whn;
        for (int off8 = threadIdx.x; off8 < NTB * H_ / 8; off8 += 128) {
            int off = off8 * 8;
            int nl  = off >> 10;             // / H_
            int k   = off & (H_ - 1);        // % H_
            v8bf v = *(const v8bf*)(wsrc + (size_t)(n0 + nl) * H_ + k);
            *(v8bf*)(&wlds[(g * NTB + nl) * LDW + k]) = v;
        }
    }
    __syncthreads();

    const float vbr = bhr[nrow], vbz = bhz[nrow], vbn = bhn[nrow];
    const bf16_t* lr = &wlds[(0 * NTB + l16) * LDW + khalf * 16];
    const bf16_t* lz = &wlds[(1 * NTB + l16) * LDW + khalf * 16];
    const bf16_t* ln = &wlds[(2 * NTB + l16) * LDW + khalf * 16];

    for (int t = 0; t < S_; ++t) {
        const bf16_t* hin   = (t & 1) ? hbuf1 : hbuf0;
        bf16_t*       hout  = (t & 1) ? hbuf0 : hbuf1;
        const float*  hfin  = (t & 1) ? hf1 : hf0;
        float*        hfout = (t & 1) ? hf0 : hf1;
        const bf16_t* arow  = hin + (size_t)(m0 + l16) * H_ + khalf * 16;

        v8f accr = {}, accz = {}, accn = {};
        for (int k0 = 0; k0 < H_; k0 += 32) {
            __builtin_prefetch(arow + k0 + 64, 0, 1);    // global_prefetch on A stream
            v16bf a = load_frag_a(arow + k0);
            accr = wmma_bf16(a, load_frag_b16(lr + k0), accr);   // B from LDS
            accz = wmma_bf16(a, load_frag_b16(lz + k0), accz);
            accn = wmma_bf16(a, load_frag_b16(ln + k0), accn);
        }

        const size_t xoff = (size_t)t * B_ * H_;
        const int mbase = m0 + 8 * khalf;
#pragma unroll
        for (int v = 0; v < 8; ++v) {
            const int m = mbase + v;
            const size_t hidx = (size_t)m * H_ + nrow;
            float xrv = (float)xr[xoff + hidx];
            float xzv = (float)xz[xoff + hidx];
            float xnv = (float)xn[xoff + hidx];
            float r  = sigmoidf_(xrv + accr[v] + vbr);
            float z  = sigmoidf_(xzv + accz[v] + vbz);
            float nn = tanhf(xnv + r * (accn[v] + vbn));
            float hp = hfin[hidx];
            float hnew = (1.0f - z) * nn + z * hp;
            hfout[hidx] = hnew;
            hout[hidx]  = (bf16_t)hnew;
            out[((size_t)m * S_ + t) * H_ + nrow] = hnew;   // outputs [B,S,H]
        }

        // ---- device-wide barrier: release h, arrive, wait for all blocks ----
        __syncthreads();
        __threadfence();
        if (threadIdx.x == 0) {
            __hip_atomic_fetch_add(&syncc[t], 1u, __ATOMIC_ACQ_REL,
                                   __HIP_MEMORY_SCOPE_AGENT);
            while (__hip_atomic_load(&syncc[t], __ATOMIC_ACQUIRE,
                                     __HIP_MEMORY_SCOPE_AGENT) < (unsigned)NBLK) {
                __builtin_amdgcn_s_sleep(2);
            }
        }
        __syncthreads();
    }
}

// ---------- final hidden state copy ----------
__global__ void copy_final(const float* __restrict__ hf, float* __restrict__ dst, int n) {
    int i = blockIdx.x * blockDim.x + threadIdx.x;
    if (i < n) dst[i] = hf[i];
}

extern "C" void kernel_launch(void* const* d_in, const int* in_sizes, int n_in,
                              void* d_out, int out_size, void* d_ws, size_t ws_size,
                              hipStream_t stream) {
    const float* inputs = (const float*)d_in[0];   // [B,S,I]
    const float* h0     = (const float*)d_in[1];   // [1,B,H]
    const float* w_ir   = (const float*)d_in[2];
    const float* w_iz   = (const float*)d_in[3];
    const float* w_in   = (const float*)d_in[4];
    const float* b_ir   = (const float*)d_in[5];
    const float* b_iz   = (const float*)d_in[6];
    const float* b_in   = (const float*)d_in[7];
    const float* w_hr   = (const float*)d_in[8];
    const float* w_hz   = (const float*)d_in[9];
    const float* w_hn   = (const float*)d_in[10];
    const float* b_hr   = (const float*)d_in[11];
    const float* b_hz   = (const float*)d_in[12];
    const float* b_hn   = (const float*)d_in[13];
    float* out = (float*)d_out;

    // workspace carve-up (all bf16 chunk element-counts are even -> f32 aligned)
    const size_t SBH = (size_t)S_ * B_ * H_;
    const size_t BSI = (size_t)B_ * S_ * I_;
    const size_t HI  = (size_t)H_ * I_;
    const size_t HH  = (size_t)H_ * H_;
    const size_t BH  = (size_t)B_ * H_;

    bf16_t* xr  = (bf16_t*)d_ws;
    bf16_t* xz  = xr  + SBH;
    bf16_t* xn  = xz  + SBH;
    bf16_t* inb = xn  + SBH;
    bf16_t* wir = inb + BSI;
    bf16_t* wiz = wir + HI;
    bf16_t* win = wiz + HI;
    bf16_t* whr = win + HI;
    bf16_t* whz = whr + HH;
    bf16_t* whn = whz + HH;
    bf16_t* hb0 = whn + HH;
    bf16_t* hb1 = hb0 + BH;
    float*  hf0 = (float*)(hb1 + BH);
    float*  hf1 = hf0 + BH;
    unsigned int* syncc = (unsigned int*)(hf1 + BH);

    // --- prep: cast everything the WMMA path consumes to bf16 ---
    cvt_f32_to_bf16<<<(int)((BSI + 255) / 256), 256, 0, stream>>>(inputs, inb, (int)BSI);
    cvt_f32_to_bf16<<<(int)((HI  + 255) / 256), 256, 0, stream>>>(w_ir, wir, (int)HI);
    cvt_f32_to_bf16<<<(int)((HI  + 255) / 256), 256, 0, stream>>>(w_iz, wiz, (int)HI);
    cvt_f32_to_bf16<<<(int)((HI  + 255) / 256), 256, 0, stream>>>(w_in, win, (int)HI);
    cvt_f32_to_bf16<<<(int)((HH  + 255) / 256), 256, 0, stream>>>(w_hr, whr, (int)HH);
    cvt_f32_to_bf16<<<(int)((HH  + 255) / 256), 256, 0, stream>>>(w_hz, whz, (int)HH);
    cvt_f32_to_bf16<<<(int)((HH  + 255) / 256), 256, 0, stream>>>(w_hn, whn, (int)HH);
    init_h<<<(int)((BH + 255) / 256), 256, 0, stream>>>(h0, hf0, hb0, (int)BH);
    zero_u32<<<(S_ + 255) / 256, 256, 0, stream>>>(syncc, S_);

    // --- phase 1: batched input projections (one WMMA GEMM, 3 gates fused) ---
    {
        dim3 grid(H_ / 64, (S_ * B_) / 32);   // 16 x 1024
        gate_gemm<<<grid, 256, 0, stream>>>(inb, wir, wiz, win,
                                            b_ir, b_iz, b_in, xr, xz, xn);
    }

    // --- phase 2: single persistent kernel runs the whole 512-step scan ---
    gru_scan<<<NBLK, 128, 0, stream>>>(whr, whz, whn, b_hr, b_hz, b_hn,
                                       xr, xz, xn, hb0, hb1, hf0, hf1,
                                       out, syncc);

    // --- final hidden state (S even -> lives in buffer 0) ---
    copy_final<<<(int)((BH + 255) / 256), 256, 0, stream>>>(hf0, out + (size_t)B_ * S_ * H_, (int)BH);
}
